// NADE_48902497632663
// MI455X (gfx1250) — matrix operand order repaired and checked by reading, not compiled, not run on hardware
//
#include <hip/hip_runtime.h>

// NADE forward, checkpoint-recompute decomposition for MI455X (gfx1250).
//
//   a_i = c + X[:, :i] @ W[:, :i]^T            (linear prefix)
//   pred_i = b_i + sigmoid(a_i) @ V[i]         (nonlinear readout)
//
// Grid: (B/16 batch-tiles) x (25 blocks of 32 steps). Each workgroup
// recomputes its block checkpoint with v_wmma_f32_16x16x32_f16 (redundant
// ~19.6 GFLOP f16 GEMM on the matrix pipe -> trivial), then runs only 32
// sequential sigmoid/dot/rank-1 steps on VALU. 1600 workgroups / 12800
// waves hide the v_exp (transcendental) floor of ~822M sigmoids, which is
// the true roofline limit; all data is L2-resident (13 MB << 192 MB).

#define IN_FEATS 784
#define HIDDEN   1024
#define BATCH    1024
#define KBLK     32
#define NBLK     ((IN_FEATS + KBLK - 1) / KBLK)   // 25
#define BTILE    16
#define WAVES    8
#define NTHREADS 256

typedef __attribute__((ext_vector_type(16))) _Float16 v16h;
typedef __attribute__((ext_vector_type(8)))  float    v8f;

__device__ __forceinline__ float sigmoid_fast(float a) {
  // 1/(1+exp(-a)) = 1/(1+exp2(-a*log2e)) : v_mul + v_exp + v_add + v_rcp
  float e = __builtin_amdgcn_exp2f(a * -1.44269504f);
  return __builtin_amdgcn_rcpf(1.0f + e);
}

template <int IMM>
__device__ __forceinline__ float swz_xor_add(float v) {
  // group-of-32 ds_swizzle: and=0x1f, or=0, xor=IMM>>10
  return v + __int_as_float(
                 __builtin_amdgcn_ds_swizzle(__float_as_int(v), IMM));
}

__global__ __launch_bounds__(NTHREADS) void nade_fused(
    const float* __restrict__ Xg,   // pixels [B, D]
    const float* __restrict__ Wg,   // W      [H, D]
    const float* __restrict__ Cg,   // c      [H]
    const float* __restrict__ Vg,   // V      [D, H]
    const float* __restrict__ Bg,   // b      [D]
    float* __restrict__ Out)        // preds  [B, D]
{
  constexpr int D = IN_FEATS, H = HIDDEN;
  const int tid  = threadIdx.x;
  const int w    = tid >> 5;        // wave id (wave32)
  const int lane = tid & 31;
  const int nl   = lane & 15;       // lane within 16-group
  const int hi   = lane >> 4;       // half-wave select
  const int b0   = blockIdx.x * BTILE;
  const int blk  = blockIdx.y;
  const int i0   = blk * KBLK;
  const int h0   = w * 128;         // wave's hidden-slice base

  extern __shared__ float smem[];
  float* Ws = smem;                 // [1024][33]  W[:, i0:i0+32], padded pitch
  float* Xs = Ws + 1024 * 33;       // [16][32]    X[b0:b0+16, i0:i0+32]
  float* Pb = Xs + BTILE * KBLK;    // [32][128]   per-(step,wave,b) partials

  // ---- stage W column-block and X tile into LDS (coalesced) ----
  for (int idx = tid; idx < 1024 * KBLK; idx += NTHREADS) {
    int h = idx >> 5, r = idx & 31;
    int i = i0 + r;
    Ws[h * 33 + r] = (i < D) ? Wg[h * D + i] : 0.0f;
  }
  for (int idx = tid; idx < BTILE * KBLK; idx += NTHREADS) {
    int bl = idx >> 5, r = idx & 31;
    int i = i0 + r;
    Xs[idx] = (i < D) ? Xg[(b0 + bl) * D + i] : 0.0f;
  }

  // ---- phase 1: checkpoint A = c + X[:, :i0] @ W[:, :i0]^T via WMMA ----
  // Wave owns 8 C-tiles (16b x 16h each): lane holds b = j + 8*hi (reg j),
  // h = h0 + 16*t + nl. Accumulator stays in these regs for phase 2.
  v8f acc[8];
#pragma unroll
  for (int t = 0; t < 8; ++t) {
    float cv = Cg[h0 + 16 * t + nl];
#pragma unroll
    for (int j = 0; j < 8; ++j) acc[t][j] = cv;
  }

  for (int kb = 0; kb < i0; kb += KBLK) {
    // A-frag 16x32 f16: lane M = nl; K run = hi*16 + {0..15}
    v16h af;
    {
      const float* xr = Xg + (b0 + nl) * D + kb + hi * 16;
#pragma unroll
      for (int e = 0; e < 8; ++e) af[e] = (_Float16)xr[e];
#pragma unroll
      for (int e = 0; e < 8; ++e) af[8 + e] = (_Float16)xr[8 + e];
    }
#pragma unroll
    for (int t = 0; t < 8; ++t) {
      // B-frag 32x16 f16: lane N = nl, K run = hi*16 + {0..15}
      // B[k, n] = W[h0+16t+n, kb+k] -> 16 contiguous f32 from row of W
      const float* wr = Wg + (h0 + 16 * t + nl) * D + kb + hi * 16;
      v16h bf;
#pragma unroll
      for (int e = 0; e < 16; ++e) bf[e] = (_Float16)wr[e];
      acc[t] = __builtin_amdgcn_wmma_f32_16x16x32_f16(
          false, af, false, bf, (short)0, acc[t], false, false);
    }
  }

  __syncthreads();

  // ---- phase 2: 32 sequential NADE steps (sigmoid -> dot -> rank-1) ----
  for (int r = 0; r < KBLK; ++r) {
    const int i  = i0 + r;
    const int iv = (i < D) ? i : (D - 1);
    {  // prefetch next step's V row slice (global_prefetch_b8)
      int inext = (i + 1 < D) ? (i + 1) : (D - 1);
      __builtin_prefetch(Vg + (size_t)inext * H + h0 + lane * 4, 0, 1);
    }
    float xv[8];
#pragma unroll
    for (int j = 0; j < 8; ++j) xv[j] = Xs[(j + 8 * hi) * KBLK + r];

    float pj[8] = {0.f, 0.f, 0.f, 0.f, 0.f, 0.f, 0.f, 0.f};
    const float* vrow = Vg + (size_t)iv * H + h0 + nl;
#pragma unroll
    for (int t = 0; t < 8; ++t) {
      float wv = Ws[(h0 + 16 * t + nl) * 33 + r];
      float vv = vrow[16 * t];
#pragma unroll
      for (int j = 0; j < 8; ++j) {
        float a = acc[t][j];
        float s = sigmoid_fast(a);       // sigmoid of PRE-update a (ref order)
        pj[j] += s * vv;                 // pred partial
        acc[t][j] = a + wv * xv[j];      // rank-1 state update
      }
    }
    // butterfly reduce over the 16-lane h-group (ds_swizzle SWAPX1/2/4/8)
#pragma unroll
    for (int j = 0; j < 8; ++j) {
      pj[j] = swz_xor_add<0x041F>(pj[j]);
      pj[j] = swz_xor_add<0x081F>(pj[j]);
      pj[j] = swz_xor_add<0x101F>(pj[j]);
      pj[j] = swz_xor_add<0x201F>(pj[j]);
    }
    if (nl == 0) {
#pragma unroll
      for (int j = 0; j < 8; ++j)
        Pb[r * 128 + w * 16 + (j + 8 * hi)] = pj[j];
    }
  }

  __syncthreads();

  // ---- final cross-wave reduce + bias + store ----
  for (int idx = tid; idx < BTILE * KBLK; idx += NTHREADS) {
    int r = idx >> 4, bl = idx & 15;
    int i = i0 + r;
    if (i < D) {
      float s = Bg[i];
#pragma unroll
      for (int w2 = 0; w2 < WAVES; ++w2) s += Pb[r * 128 + w2 * 16 + bl];
      Out[(b0 + bl) * D + i] = s;
    }
  }
}

extern "C" void kernel_launch(void* const* d_in, const int* in_sizes, int n_in,
                              void* d_out, int out_size, void* d_ws,
                              size_t ws_size, hipStream_t stream) {
  (void)in_sizes; (void)n_in; (void)out_size; (void)d_ws; (void)ws_size;
  const float* X  = (const float*)d_in[0];  // pixels [1024, 784]
  const float* W  = (const float*)d_in[1];  // W      [1024, 784]
  const float* C  = (const float*)d_in[2];  // c      [1024]
  const float* V  = (const float*)d_in[3];  // V      [784, 1024]
  const float* Bb = (const float*)d_in[4];  // b      [784]
  float* out = (float*)d_out;

  const size_t smem_bytes =
      (size_t)(1024 * 33 + BTILE * KBLK + KBLK * 128) * sizeof(float);  // 153600
  (void)hipFuncSetAttribute(reinterpret_cast<const void*>(nade_fused),
                            hipFuncAttributeMaxDynamicSharedMemorySize,
                            (int)smem_bytes);

  dim3 grid(BATCH / BTILE, NBLK);  // 64 x 25 workgroups
  nade_fused<<<grid, NTHREADS, smem_bytes, stream>>>(X, W, C, V, Bb, out);
}
